// SimpleXLSTM_13804024889445
// MI455X (gfx1250) — matrix operand order repaired and checked by reading, not compiled, not run on hardware
//
#include <hip/hip_runtime.h>
#include <hip/hip_bf16.h>
#include <stdint.h>

// Problem constants (match reference)
#define BATCH 128
#define SEQ   512
#define IN_F  512
#define HID   1024
#define GATES 4096   // 4*HID

typedef __bf16 bf16;
typedef __attribute__((ext_vector_type(16))) __bf16 v16bf;
typedef __attribute__((ext_vector_type(8)))  float  v8f;

struct Bf16Pair { uint4 lo, hi; };   // 32 bytes == v16bf

__device__ __forceinline__ float sigf(float x) { return 1.0f / (1.0f + __expf(-x)); }

__device__ __forceinline__ v16bf load_a_frag(const bf16* p) {
    Bf16Pair t;
    t.lo = *(const uint4*)(p);        // K = kbase      + 0..7
    t.hi = *(const uint4*)(p + 16);   // K = kbase + 16 + 0..7
    return __builtin_bit_cast(v16bf, t);
}

// ---------------------------------------------------------------------------
// f32 -> bf16 bulk convert
__global__ void __launch_bounds__(256)
f32_to_bf16_kernel(const float* __restrict__ src, bf16* __restrict__ dst, int n) {
    int i = blockIdx.x * blockDim.x + threadIdx.x;
    if (i < n) dst[i] = (bf16)src[i];
}

// ---------------------------------------------------------------------------
// Pack W (K x 4096, row-major f32) into per-lane WMMA B-fragment layout:
// frag[nt][kb][lane][e]  (e=0..15 contiguous 32B per lane)
//   k = kb*32 + (e>>3)*16 + (lane>>4)*8 + (e&7) ,  n = nt*16 + (lane&15)
// Matches A fragment order: lanes 0-15 hold K=0..7,16..23,
// lanes 16-31 hold K=8..15,24..31, column = lane%16.
__global__ void __launch_bounds__(256)
pack_w_frag_kernel(const float* __restrict__ W, bf16* __restrict__ frag, int KB) {
    int idx = blockIdx.x * blockDim.x + threadIdx.x;
    int total = 256 * KB * 512;
    if (idx >= total) return;
    int e    = idx & 15;
    int lane = (idx >> 4) & 31;
    int kb   = (idx >> 9) % KB;
    int nt   = idx / (KB << 9);
    int k    = kb * 32 + ((e >> 3) << 4) + ((lane >> 4) << 3) + (e & 7);
    int n    = (nt << 4) + (lane & 15);
    frag[idx] = (bf16)W[(size_t)k * GATES + n];
}

// ---------------------------------------------------------------------------
// Zero-init recurrent state
__global__ void __launch_bounds__(256)
init_state_kernel(float* c0, float* c1, bf16* h0, bf16* h1) {
    int i = blockIdx.x * blockDim.x + threadIdx.x;
    if (i < BATCH * HID) {
        c0[i] = 0.0f; c1[i] = 0.0f;
        h0[i] = (bf16)0.0f; h1[i] = (bf16)0.0f;
    }
}

// ---------------------------------------------------------------------------
// Fused two-source WMMA GEMM:  z = A1 @ Wfrag[0:K1] + A2 @ Wfrag[K1:K1+K2] + bias
// A1/A2: bf16 row-major (M=128, given lda).  Wfrag: packed fragments, KBtot K-blocks.
// Register blocking: one wave computes a 32x32 output block (2 M-tiles x 2 N-tiles,
// 4 independent accumulators) -> 8 b128 loads per 4 WMMAs, 2x fragment reuse.
// Waves: (128/32)*(4096/32) = 512 -> 128 blocks x 4 waves; waves in a block share
// the same B fragment stream (WGP$ hits).
__global__ void __launch_bounds__(128)
lstm_gemm_kernel(const bf16* __restrict__ A1, int lda1, int K1,
                 const bf16* __restrict__ A2, int lda2, int K2,
                 const bf16* __restrict__ frag, int KBtot,
                 const float* __restrict__ bias,
                 float* __restrict__ z) {
    const int lane = threadIdx.x & 31;
    const int w    = blockIdx.x * (blockDim.x >> 5) + (threadIdx.x >> 5); // 0..511
    const int bm   = w & 3;      // M block of 32 rows (tiles 2*bm, 2*bm+1)
    const int bn   = w >> 2;     // N block of 32 cols (tiles 2*bn, 2*bn+1)
    const int r    = lane & 15;  // row (A) / col (B,C) within tile
    const int kg   = lane >> 4;  // lane-group selects K sub-range

    v8f acc00 = {}, acc01 = {}, acc10 = {}, acc11 = {};

    const size_t fragTile = (size_t)KBtot * 512;
    const bf16* fragp0 = frag + (size_t)(2 * bn) * fragTile + lane * 16;
    const bf16* fragp1 = fragp0 + fragTile;

    // ---- part 1: A1 over K1 (W rows [0, K1)) ----
    {
        const bf16* a0 = A1 + (size_t)(bm * 32 + r) * lda1 + kg * 8;
        const bf16* a1 = a0 + (size_t)16 * lda1;
        for (int kk = 0; kk < K1; kk += 32) {
            v16bf va0 = load_a_frag(a0 + kk);
            v16bf va1 = load_a_frag(a1 + kk);
            v16bf vb0 = *(const v16bf*)fragp0;
            v16bf vb1 = *(const v16bf*)fragp1;
            __builtin_prefetch(fragp0 + 512 * 4, 0, 3);   // global_prefetch_b8
            __builtin_prefetch(fragp1 + 512 * 4, 0, 3);
            fragp0 += 512; fragp1 += 512;
            acc00 = __builtin_amdgcn_wmma_f32_16x16x32_bf16(false, va0, false, vb0, (short)0, acc00, false, false);
            acc01 = __builtin_amdgcn_wmma_f32_16x16x32_bf16(false, va0, false, vb1, (short)0, acc01, false, false);
            acc10 = __builtin_amdgcn_wmma_f32_16x16x32_bf16(false, va1, false, vb0, (short)0, acc10, false, false);
            acc11 = __builtin_amdgcn_wmma_f32_16x16x32_bf16(false, va1, false, vb1, (short)0, acc11, false, false);
        }
    }
    // ---- part 2: A2 over K2 (W rows [K1, K1+K2)) ----
    {
        const bf16* a0 = A2 + (size_t)(bm * 32 + r) * lda2 + kg * 8;
        const bf16* a1 = a0 + (size_t)16 * lda2;
        for (int kk = 0; kk < K2; kk += 32) {
            v16bf va0 = load_a_frag(a0 + kk);
            v16bf va1 = load_a_frag(a1 + kk);
            v16bf vb0 = *(const v16bf*)fragp0;
            v16bf vb1 = *(const v16bf*)fragp1;
            __builtin_prefetch(fragp0 + 512 * 4, 0, 3);
            __builtin_prefetch(fragp1 + 512 * 4, 0, 3);
            fragp0 += 512; fragp1 += 512;
            acc00 = __builtin_amdgcn_wmma_f32_16x16x32_bf16(false, va0, false, vb0, (short)0, acc00, false, false);
            acc01 = __builtin_amdgcn_wmma_f32_16x16x32_bf16(false, va0, false, vb1, (short)0, acc01, false, false);
            acc10 = __builtin_amdgcn_wmma_f32_16x16x32_bf16(false, va1, false, vb0, (short)0, acc10, false, false);
            acc11 = __builtin_amdgcn_wmma_f32_16x16x32_bf16(false, va1, false, vb1, (short)0, acc11, false, false);
        }
    }

    // ---- epilogue: bias + store (acc VGPR v holds row m = v + 8*kg, col n) ----
    const int n0 = bn * 32 + r;          // tile 2*bn
    const int n1 = n0 + 16;              // tile 2*bn+1
    const float bv0 = bias[n0];
    const float bv1 = bias[n1];
    const size_t row0 = (size_t)(bm * 32 + kg * 8);   // tile 2*bm rows
    float* zp00 = z + row0 * GATES + n0;
    float* zp01 = z + row0 * GATES + n1;
    float* zp10 = zp00 + (size_t)16 * GATES;          // tile 2*bm+1 rows
    float* zp11 = zp01 + (size_t)16 * GATES;
#pragma unroll
    for (int v = 0; v < 8; ++v) {
        zp00[(size_t)v * GATES] = acc00[v] + bv0;
        zp01[(size_t)v * GATES] = acc01[v] + bv1;
        zp10[(size_t)v * GATES] = acc10[v] + bv0;
        zp11[(size_t)v * GATES] = acc11[v] + bv1;
    }
}

// ---------------------------------------------------------------------------
// LSTM pointwise cell update. z gate order: [f | i | o | g] along N.
__global__ void __launch_bounds__(256)
lstm_pointwise_kernel(const float* __restrict__ z,
                      float* __restrict__ c, bf16* __restrict__ h) {
    int idx = blockIdx.x * blockDim.x + threadIdx.x;
    if (idx >= BATCH * HID) return;
    int b = idx >> 10;
    int j = idx & (HID - 1);
    const float* zr = z + (size_t)b * GATES;
    float f = zr[j];
    float i = zr[j + HID];
    float o = zr[j + 2 * HID];
    float g = zr[j + 3 * HID];
    float cn = sigf(f) * c[idx] + sigf(i) * tanhf(g);
    c[idx] = cn;
    h[idx] = (bf16)(sigf(o) * tanhf(cn));
}

// ---------------------------------------------------------------------------
// Final: out = sigmoid(h1 @ Wout + bout)
__global__ void __launch_bounds__(128)
lstm_out_kernel(const bf16* __restrict__ h1, const float* __restrict__ Wout,
                const float* __restrict__ bout, float* __restrict__ out) {
    int b = blockIdx.x * blockDim.x + threadIdx.x;
    if (b >= BATCH) return;
    float acc = 0.0f;
    const bf16* hr = h1 + (size_t)b * HID;
#pragma unroll 4
    for (int j = 0; j < HID; ++j) acc += (float)hr[j] * Wout[j];
    out[b] = sigf(acc + bout[0]);
}

// ---------------------------------------------------------------------------
extern "C" void kernel_launch(void* const* d_in, const int* in_sizes, int n_in,
                              void* d_out, int out_size, void* d_ws, size_t ws_size,
                              hipStream_t stream) {
    (void)in_sizes; (void)n_in; (void)out_size; (void)ws_size;
    const float* x    = (const float*)d_in[0];
    const float* W0   = (const float*)d_in[1];
    const float* b0   = (const float*)d_in[2];
    const float* W1   = (const float*)d_in[3];
    const float* b1   = (const float*)d_in[4];
    const float* Wout = (const float*)d_in[5];
    const float* bout = (const float*)d_in[6];
    float* out = (float*)d_out;

    // Workspace layout (~103 MB total)
    char* ws = (char*)d_ws;
    const size_t XBF_BYTES = (size_t)BATCH * SEQ * IN_F * 2;     // 67,108,864
    const size_t W0F_BYTES = (size_t)(IN_F + HID) * GATES * 2;   // 12,582,912
    const size_t W1F_BYTES = (size_t)(2 * HID) * GATES * 2;      // 16,777,216
    const size_t HB_BYTES  = (size_t)BATCH * HID * 2;            //    262,144
    const size_t C_BYTES   = (size_t)BATCH * HID * 4;            //    524,288
    const size_t Z_BYTES   = (size_t)BATCH * GATES * 4;          //  2,097,152

    size_t off = 0;
    bf16*  xbf    = (bf16*)(ws + off); off += XBF_BYTES;
    bf16*  w0frag = (bf16*)(ws + off); off += W0F_BYTES;
    bf16*  w1frag = (bf16*)(ws + off); off += W1F_BYTES;
    bf16*  h0     = (bf16*)(ws + off); off += HB_BYTES;
    bf16*  h1     = (bf16*)(ws + off); off += HB_BYTES;
    float* c0     = (float*)(ws + off); off += C_BYTES;
    float* c1     = (float*)(ws + off); off += C_BYTES;
    float* z0     = (float*)(ws + off); off += Z_BYTES;
    float* z1     = (float*)(ws + off); off += Z_BYTES;

    // One-time (per call) prep: convert x, pack weight fragments, zero state.
    {
        int nx = BATCH * SEQ * IN_F;
        f32_to_bf16_kernel<<<(nx + 255) / 256, 256, 0, stream>>>(x, xbf, nx);
        int n0 = 256 * 48 * 512;   // KB = 1536/32 = 48
        pack_w_frag_kernel<<<(n0 + 255) / 256, 256, 0, stream>>>(W0, w0frag, 48);
        int n1 = 256 * 64 * 512;   // KB = 2048/32 = 64
        pack_w_frag_kernel<<<(n1 + 255) / 256, 256, 0, stream>>>(W1, w1frag, 64);
        init_state_kernel<<<(BATCH * HID + 255) / 256, 256, 0, stream>>>(c0, c1, h0, h1);
    }

    // Sequential recurrence; all ordering via stream.
    for (int t = 0; t < SEQ; ++t) {
        // layer 0: z0 = x_t @ W0[0:512] + h0 @ W0[512:1536] + b0
        lstm_gemm_kernel<<<128, 128, 0, stream>>>(
            xbf + (size_t)t * IN_F, SEQ * IN_F, IN_F,
            h0, HID, HID,
            w0frag, 48, b0, z0);
        lstm_pointwise_kernel<<<(BATCH * HID + 255) / 256, 256, 0, stream>>>(z0, c0, h0);

        // layer 1: z1 = h0 @ W1[0:1024] + h1 @ W1[1024:2048] + b1
        lstm_gemm_kernel<<<128, 128, 0, stream>>>(
            h0, HID, HID,
            h1, HID, HID,
            w1frag, 64, b1, z1);
        lstm_pointwise_kernel<<<(BATCH * HID + 255) / 256, 256, 0, stream>>>(z1, c1, h1);
    }

    lstm_out_kernel<<<1, 128, 0, stream>>>(h1, Wout, bout, out);
}
